// DrugEncoder_12025908429009
// MI455X (gfx1250) — compile-verified, matrix operand order checked by reading, not run on hardware
//
#include <hip/hip_runtime.h>
#include <hip/hip_bf16.h>
#include <math.h>

#define HIDDEN 256
#define LN_EPS 1e-5f

typedef __attribute__((ext_vector_type(16))) _Float16 v16h;
typedef __attribute__((ext_vector_type(8)))  float    v8f;

union V16H { v16h v; _Float16 e[16]; };
union V8F  { v8f  v; float    e[8];  };

__device__ __forceinline__ float silu_f(float v) { return v / (1.0f + __expf(-v)); }

// ---------------------------------------------------------------------------
// Convert fp32 weights [K,256] -> f16 in WMMA-fragment-swizzled order:
//   dst[((kb*16 + nt)*32 + lane)*16 + j] = src[k, nt*16 + (lane&15)]
//   with k = kb*32 + ((j>>3)<<4) + ((lane>>4)<<3) + (j&7)
// so a B fragment is 16 contiguous f16 per lane (two ds_load_b128).
// ---------------------------------------------------------------------------
__global__ void convert_w_swizzle_kernel(const float* __restrict__ Wsrc,
                                         _Float16* __restrict__ Wdst, int nElem) {
    int o = blockIdx.x * blockDim.x + threadIdx.x;
    if (o >= nElem) return;
    int j    = o & 15;
    int lane = (o >> 4) & 31;
    int nt   = (o >> 9) & 15;
    int kb   = o >> 13;
    int k    = kb * 32 + ((j >> 3) << 4) + ((lane >> 4) << 3) + (j & 7);
    int col  = nt * 16 + (lane & 15);
    Wdst[o]  = (_Float16)Wsrc[(size_t)k * HIDDEN + col];
}

// ---------------------------------------------------------------------------
// agg = h  (float4 copy)
// ---------------------------------------------------------------------------
__global__ void copy_f4_kernel(const float4* __restrict__ src, float4* __restrict__ dst, int n4) {
    int i = blockIdx.x * blockDim.x + threadIdx.x;
    if (i < n4) dst[i] = src[i];
}

// ---------------------------------------------------------------------------
// agg[dst] += relu(h[src])   -- 64 threads (2 waves) per edge, float4 lanes
// ---------------------------------------------------------------------------
__global__ void edge_scatter_kernel(const float* __restrict__ h, float* agg,
                                    const int* __restrict__ src, const int* __restrict__ dst,
                                    int E) {
    long long t = (long long)blockIdx.x * blockDim.x + threadIdx.x;
    int e = (int)(t >> 6);
    if (e >= E) return;
    int c = ((int)t & 63) * 4;
    int s = src[e], d = dst[e];
    float4 m = *reinterpret_cast<const float4*>(h + (size_t)s * HIDDEN + c);
    float* o = agg + (size_t)d * HIDDEN + c;
    atomicAdd(o + 0, fmaxf(m.x, 0.0f));
    atomicAdd(o + 1, fmaxf(m.y, 0.0f));
    atomicAdd(o + 2, fmaxf(m.z, 0.0f));
    atomicAdd(o + 3, fmaxf(m.w, 0.0f));
}

// ---------------------------------------------------------------------------
// WMMA GEMM: out[M,256] = epi(A[M,KDIM] @ B + bias), B16sw pre-swizzled f16.
//   LNRES=false : epi = SiLU                       (input projection)
//   LNRES=true  : epi = SiLU(LayerNorm(.)) + resid (GNN layer, in-place h)
// Block: 256 threads = 8 waves, tile 64 x 256.
// Wave w: m-tile = w>>1 (16 rows), n-half = w&1 (8 n-tiles of 16 cols).
// ---------------------------------------------------------------------------
template <int KDIM, bool LNRES>
__global__ __launch_bounds__(256)
void gemm_tile_kernel(const float* __restrict__ A, const _Float16* __restrict__ B16sw,
                      const float* __restrict__ bias,
                      const float* __restrict__ gamma, const float* __restrict__ beta,
                      const float* resid, float* out, int Mtot) {
    __shared__ _Float16 As[64 * 32];        //  4 KB, A tile f32->f16 (row-major [row][32])
    __shared__ _Float16 Bs[16 * 32 * 16];   // 16 KB, B tile in fragment order [nt][lane][16]
    __shared__ float    red[2 * 64];        // LN row sums / sq-sums
    __shared__ float    stat[2 * 64];       // LN mu / rstd

    const int tid   = threadIdx.x;
    const int lane  = tid & 31;
    const int wave  = tid >> 5;
    const int l16   = lane & 15;
    const int lhalf = lane >> 4;
    const int mtile = wave >> 1;
    const int nhalf = wave & 1;
    const int rowBase = blockIdx.x * 64;

    V8F acc[8];
#pragma unroll
    for (int t = 0; t < 8; t++)
#pragma unroll
        for (int r = 0; r < 8; r++) acc[t].e[r] = 0.0f;

    for (int k0 = 0; k0 < KDIM; k0 += 32) {
        __syncthreads();
        // ---- stage A tile (64x32 f32 -> f16 in LDS) ----
#pragma unroll
        for (int i = 0; i < 2; i++) {
            int idx = tid + i * 256;               // 0..511 -> 512 float4
            int row = idx >> 3, seg = idx & 7;
            int grow = rowBase + row;
            float4 a4 = make_float4(0.f, 0.f, 0.f, 0.f);
            if (grow < Mtot)
                a4 = *reinterpret_cast<const float4*>(A + (size_t)grow * KDIM + k0 + seg * 4);
            _Float16* p = &As[row * 32 + seg * 4];
            p[0] = (_Float16)a4.x; p[1] = (_Float16)a4.y;
            p[2] = (_Float16)a4.z; p[3] = (_Float16)a4.w;
        }
        // ---- stage B tile: contiguous 8192-f16 chunk for this k-block ----
        {
            const float4* bsrc =
                reinterpret_cast<const float4*>(B16sw + (size_t)(k0 >> 5) * (32 * HIDDEN));
            float4* bdst = reinterpret_cast<float4*>(Bs);
#pragma unroll
            for (int i = 0; i < 4; i++) bdst[tid + i * 256] = bsrc[tid + i * 256];
        }
        __syncthreads();

        // ---- A fragment: two contiguous b128 LDS loads ----
        V16H a;
        {
            const _Float16* ap = &As[(mtile * 16 + l16) * 32 + lhalf * 8];
            *reinterpret_cast<float4*>(&a.e[0]) = *reinterpret_cast<const float4*>(ap);
            *reinterpret_cast<float4*>(&a.e[8]) = *reinterpret_cast<const float4*>(ap + 16);
        }
        // ---- 8 n-tiles: B fragment (two b128) + WMMA ----
#pragma unroll
        for (int t = 0; t < 8; t++) {
            V16H bfrag;
            const _Float16* bp = &Bs[(((nhalf * 8 + t) * 32) + lane) * 16];
            *reinterpret_cast<float4*>(&bfrag.e[0]) = *reinterpret_cast<const float4*>(bp);
            *reinterpret_cast<float4*>(&bfrag.e[8]) = *reinterpret_cast<const float4*>(bp + 8);
            acc[t].v = __builtin_amdgcn_wmma_f32_16x16x32_f16(
                false, a.v, false, bfrag.v, (short)0, acc[t].v, false, false);
        }
    }

    // ---- epilogue ----
    if (!LNRES) {
#pragma unroll
        for (int t = 0; t < 8; t++) {
            int col = (nhalf * 8 + t) * 16 + l16;
            float bv = bias[col];
#pragma unroll
            for (int r = 0; r < 8; r++) {
                int grow = rowBase + mtile * 16 + lhalf * 8 + r;
                if (grow < Mtot)
                    out[(size_t)grow * HIDDEN + col] = silu_f(acc[t].e[r] + bv);
            }
        }
    } else {
        __syncthreads();
        if (tid < 64) { red[tid] = 0.0f; red[64 + tid] = 0.0f; }
        __syncthreads();

        float bv[8];
#pragma unroll
        for (int t = 0; t < 8; t++) bv[t] = bias[(nhalf * 8 + t) * 16 + l16];

#pragma unroll
        for (int r = 0; r < 8; r++) {
            float s = 0.0f, q = 0.0f;
#pragma unroll
            for (int t = 0; t < 8; t++) {
                float v = acc[t].e[r] + bv[t];
                s += v; q += v * v;
            }
            int rl = mtile * 16 + lhalf * 8 + r;
            atomicAdd(&red[rl], s);        // ds_add_f32
            atomicAdd(&red[64 + rl], q);
        }
        __syncthreads();
        if (tid < 64) {
            float mu  = red[tid] * (1.0f / HIDDEN);
            float var = red[64 + tid] * (1.0f / HIDDEN) - mu * mu;
            stat[tid]      = mu;
            stat[64 + tid] = rsqrtf(var + LN_EPS);
        }
        __syncthreads();
#pragma unroll
        for (int t = 0; t < 8; t++) {
            int col = (nhalf * 8 + t) * 16 + l16;
            float g = gamma[col], bt = beta[col];
#pragma unroll
            for (int r = 0; r < 8; r++) {
                int rl = mtile * 16 + lhalf * 8 + r;
                int grow = rowBase + rl;
                if (grow < Mtot) {
                    float v  = acc[t].e[r] + bv[t];
                    float ln = g * (v - stat[rl]) * stat[64 + rl] + bt;
                    size_t off = (size_t)grow * HIDDEN + col;
                    out[off] = silu_f(ln) + resid[off];   // resid aliases out (read-then-write)
                }
            }
        }
    }
}

// ---------------------------------------------------------------------------
// Global mean pool
// ---------------------------------------------------------------------------
__global__ void zero_kernel(float* p, int n) {
    int i = blockIdx.x * blockDim.x + threadIdx.x;
    if (i < n) p[i] = 0.0f;
}

__global__ void pool_accum_kernel(const float* __restrict__ h, const int* __restrict__ batch,
                                  float* sums, float* cnt, int N) {
    long long t = (long long)blockIdx.x * blockDim.x + threadIdx.x;
    int n = (int)(t >> 6);
    if (n >= N) return;
    int c = ((int)t & 63) * 4;
    int g = batch[n];
    float4 v = *reinterpret_cast<const float4*>(h + (size_t)n * HIDDEN + c);
    float* o = sums + (size_t)g * HIDDEN + c;
    atomicAdd(o + 0, v.x); atomicAdd(o + 1, v.y);
    atomicAdd(o + 2, v.z); atomicAdd(o + 3, v.w);
    if (((int)t & 63) == 0) atomicAdd(&cnt[g], 1.0f);
}

__global__ void pool_div_kernel(float* out, const float* __restrict__ cnt, int n) {
    int i = blockIdx.x * blockDim.x + threadIdx.x;
    if (i < n) out[i] /= fmaxf(cnt[i >> 8], 1.0f);   // i>>8 == i / HIDDEN
}

// ---------------------------------------------------------------------------
extern "C" void kernel_launch(void* const* d_in, const int* in_sizes, int n_in,
                              void* d_out, int out_size, void* d_ws, size_t ws_size,
                              hipStream_t stream) {
    const float* x     = (const float*)d_in[0];
    const int*   edge  = (const int*)d_in[1];   // [2, E]
    const int*   batch = (const int*)d_in[2];
    const float* Win   = (const float*)d_in[3];
    const float* b_in  = (const float*)d_in[4];
    const float* W     = (const float*)d_in[5];
    const float* b     = (const float*)d_in[6];
    const float* gamma = (const float*)d_in[7];
    const float* beta  = (const float*)d_in[8];
    float* out = (float*)d_out;

    const int NODE_IN = 64, L = 3;
    const int N = in_sizes[0] / NODE_IN;
    const int E = in_sizes[1] / 2;
    const int G = out_size / HIDDEN;

    // scratch layout (256-byte aligned slices)
    char* ws = (char*)d_ws;
    size_t off = 0;
    auto carve = [&](size_t bytes) -> void* {
        void* p = ws + off;
        off = (off + bytes + 255) & ~(size_t)255;
        return p;
    };
    float*    h     = (float*)carve((size_t)N * HIDDEN * sizeof(float));
    float*    agg   = (float*)carve((size_t)N * HIDDEN * sizeof(float));
    _Float16* Win16 = (_Float16*)carve((size_t)NODE_IN * HIDDEN * sizeof(_Float16));
    _Float16* W16   = (_Float16*)carve((size_t)L * HIDDEN * HIDDEN * sizeof(_Float16));
    float*    cnt   = (float*)carve((size_t)G * sizeof(float));
    (void)ws_size; (void)n_in;

    // 1. weights -> f16, pre-swizzled to WMMA fragment order
    int nWin = NODE_IN * HIDDEN;
    int nWl  = HIDDEN * HIDDEN;
    convert_w_swizzle_kernel<<<(nWin + 255) / 256, 256, 0, stream>>>(Win, Win16, nWin);
    for (int l = 0; l < L; l++)
        convert_w_swizzle_kernel<<<(nWl + 255) / 256, 256, 0, stream>>>(
            W + (size_t)l * nWl, W16 + (size_t)l * nWl, nWl);

    // 2. h = silu(x @ W_in + b_in)
    int mblocks = (N + 63) / 64;
    gemm_tile_kernel<NODE_IN, false><<<mblocks, 256, 0, stream>>>(
        x, Win16, b_in, nullptr, nullptr, nullptr, h, N);

    // 3. GNN layers
    int n4 = N * HIDDEN / 4;
    long long escatter = (long long)E * 64;
    for (int l = 0; l < L; l++) {
        copy_f4_kernel<<<(n4 + 255) / 256, 256, 0, stream>>>((const float4*)h, (float4*)agg, n4);
        edge_scatter_kernel<<<(int)((escatter + 255) / 256), 256, 0, stream>>>(
            h, agg, edge, edge + E, E);
        gemm_tile_kernel<HIDDEN, true><<<mblocks, 256, 0, stream>>>(
            agg, W16 + (size_t)l * nWl,
            b + l * HIDDEN, gamma + l * HIDDEN, beta + l * HIDDEN, h, h, N);
    }

    // 4. global mean pool
    int nz = G * HIDDEN;
    zero_kernel<<<(nz + 255) / 256, 256, 0, stream>>>(out, nz);
    zero_kernel<<<(G + 255) / 256, 256, 0, stream>>>(cnt, G);
    long long npool = (long long)N * 64;
    pool_accum_kernel<<<(int)((npool + 255) / 256), 256, 0, stream>>>(h, batch, out, cnt, N);
    pool_div_kernel<<<(nz + 255) / 256, 256, 0, stream>>>(out, cnt, nz);
}
